// backbone_HGE_HGA_1228360647125
// MI455X (gfx1250) — matrix operand order repaired and checked by reading, not compiled, not run on hardware
//
#include <hip/hip_runtime.h>
#include <hip/hip_bf16.h>

// ---------------------------------------------------------------------------
// Types
// ---------------------------------------------------------------------------
typedef __attribute__((ext_vector_type(8)))  float        v8f;
typedef __attribute__((ext_vector_type(16))) __bf16       v16bf;
typedef __attribute__((ext_vector_type(4)))  unsigned int v4u;
typedef __attribute__((ext_vector_type(4)))  int          v4i;

// CDNA5 async global->LDS copy (guarded; fallback = plain 16B copy)
#if defined(__has_builtin)
#if __has_builtin(__builtin_amdgcn_global_load_async_to_lds_b128) && \
    __has_builtin(__builtin_amdgcn_s_wait_asynccnt)
#define HAVE_ASYNC_LDS 1
#endif
#endif

__device__ __forceinline__ void copy16_g2l(const unsigned short* g, unsigned short* l) {
#if defined(HAVE_ASYNC_LDS)
    __builtin_amdgcn_global_load_async_to_lds_b128(
        (__attribute__((address_space(1))) v4i*)(unsigned long long)(g),
        (__attribute__((address_space(3))) v4i*)(unsigned int)(unsigned long long)(l),
        0, 0);
#else
    *(v4u*)l = *(const v4u*)g;
#endif
}
__device__ __forceinline__ void async_wait_all() {
#if defined(HAVE_ASYNC_LDS)
    __builtin_amdgcn_s_wait_asynccnt(0);
#endif
}

// ---------------------------------------------------------------------------
// Constant stencils (5x5; Sobel 3x3 zero-padded into center)
// ---------------------------------------------------------------------------
__device__ __constant__ float c_GL[25] = {
    0.f,        0.f,        -1.f/16.f,  0.f,        0.f,
    0.f,        -1.f/16.f,  -1.f/8.f,   -1.f/16.f,  0.f,
    -1.f/16.f,  -1.f/8.f,    1.f,       -1.f/8.f,   -1.f/16.f,
    0.f,        -1.f/16.f,  -1.f/8.f,   -1.f/16.f,  0.f,
    0.f,        0.f,        -1.f/16.f,  0.f,        0.f};
__device__ __constant__ float c_S0[25] = {
    0,0,0,0,0,  0,1,2,1,0,  0,0,0,0,0,  0,-1,-2,-1,0,  0,0,0,0,0};
__device__ __constant__ float c_S45[25] = {
    0,0,0,0,0,  0,2,1,0,0,  0,1,0,-1,0,  0,0,-1,-2,0,  0,0,0,0,0};
__device__ __constant__ float c_S90[25] = {
    0,0,0,0,0,  0,1,0,-1,0,  0,2,0,-2,0,  0,1,0,-1,0,  0,0,0,0,0};
__device__ __constant__ float c_S135[25] = {
    0,0,0,0,0,  0,0,-1,-2,0,  0,1,0,-1,0,  0,2,1,0,0,  0,0,0,0,0};

__device__ __forceinline__ unsigned short f2bf(float f) {
    unsigned u = __float_as_uint(f);
    unsigned r = u + 0x7FFFu + ((u >> 16) & 1u);   // round-to-nearest-even
    return (unsigned short)(r >> 16);
}
__device__ __forceinline__ float xval(int t) {
    float s = (float)t * 0.01f;
    return 10.f * s * s * s;
}
// A-matrix (16x32 bf16) per-lane element e -> K index (contiguous: k = 16*(lane/16) + e)
__device__ __forceinline__ int a_kidx(int lane, int e) {
    int hi = (lane >= 16);
    return (hi ? 16 : 0) + e;
}

// ===========================================================================
// init small accumulators
// ===========================================================================
__global__ void k_zero(float* att_sum, unsigned* hist, int* xmax_bits) {
    int t = threadIdx.x;
    if (t < 40) att_sum[t] = 0.f;
    if (t < 8)  xmax_bits[t] = 0;
    for (int i = t; i < 8 * 102; i += blockDim.x) hist[i] = 0u;
}

// ===========================================================================
// feat [8,64,128,128] f32 NCHW -> bf16 NHWC  ([b][h][w][ch])
// LDS-staged 64x64 transpose, coalesced on both global sides.
// ===========================================================================
__global__ __launch_bounds__(256) void k_feat_t(const float* __restrict__ feat,
                                                unsigned short* __restrict__ ft) {
    __shared__ unsigned short t[64 * 65];   // +1 pad vs bank conflicts
    int wt = blockIdx.x & 1;
    int h  = (blockIdx.x >> 1) & 127;
    int b  = blockIdx.x >> 8;
    int w0 = wt * 64;
    for (int i = threadIdx.x; i < 4096; i += 256) {
        int ch = i >> 6, w = i & 63;        // consecutive tid -> consecutive w
        t[ch * 65 + w] = f2bf(feat[((b * 64 + ch) * 128 + h) * 128 + w0 + w]);
    }
    __syncthreads();
    for (int i = threadIdx.x; i < 4096; i += 256) {
        int w = i >> 6, ch = i & 63;        // consecutive tid -> consecutive ch
        ft[((size_t)((b * 128 + h) * 128) + w0 + w) * 64 + ch] = t[ch * 65 + w];
    }
}

// ===========================================================================
// Prep: M_d = FC_d @ W_d, then combined 5x5 kernel -> WMMA A-fragments.
// Chunk ci = tap*2 + chalf  (K = 32 channels at fixed tap).
// Afrag layout: [ci 50][mb 4][lane 32][e 16]  (bf16)
// ===========================================================================
__global__ __launch_bounds__(256) void k_prep(const float* __restrict__ hw,   // [64,64,3,3]
                                              const float* __restrict__ fcw,  // [64,320]
                                              float* __restrict__ Md,         // [5,64,64]
                                              unsigned short* __restrict__ afrag) {
    const int doff[5] = {0, 1, 2, 3, 8};   // (0,0)(0,1)(0,2)(1,0)(2,2)
    int tid = threadIdx.x;
    for (int idx = tid; idx < 5 * 64 * 64; idx += 256) {
        int d = idx >> 12, o = (idx >> 6) & 63, i = idx & 63;
        const float* fc = fcw + o * 320 + d * 64;
        float s = 0.f;
        for (int c = 0; c < 64; ++c)
            s += fc[c] * hw[(c * 64 + i) * 9 + doff[d]];
        if (d == 4) s *= 4.f;
        Md[idx] = s;
    }
    __syncthreads();
    for (int idx = tid; idx < 50 * 4 * 32 * 16; idx += 256) {
        int e = idx & 15, lane = (idx >> 4) & 31, mb = (idx >> 9) & 3, ci = idx >> 11;
        int t = ci >> 1, chalf = ci & 1;
        int ch = chalf * 32 + a_kidx(lane, e);
        int o = mb * 16 + (lane & 15);
        float s = Md[(0 * 64 + o) * 64 + ch] * c_S0[t]
                + Md[(1 * 64 + o) * 64 + ch] * c_S45[t]
                + Md[(2 * 64 + o) * 64 + ch] * c_S90[t]
                + Md[(3 * 64 + o) * 64 + ch] * c_S135[t]
                + Md[(4 * 64 + o) * 64 + ch] * c_GL[t];
        afrag[idx] = f2bf(s);
    }
}

// ===========================================================================
// HGE main GEMM (implicit 64->64 5x5 conv).
// Block = (b, h-pair): LDS tile [6 rows][132 cols][64 ch] bf16 (NHWC-sliced),
// loaded with async global->LDS b128 copies. 8 waves = 2 rows x 4 quadrants;
// each wave: 32 cols (two 16-wide N tiles sharing A), all 64 out channels.
// Per chunk: 2x v16bf LDS loads + 4x v16bf A loads -> 8x v_wmma.
// ===========================================================================
__global__ __launch_bounds__(256) void k_hge_gemm(const unsigned short* __restrict__ feat_t,
                                                  const unsigned short* __restrict__ afrag,
                                                  const float* __restrict__ fcb,
                                                  float* __restrict__ outp) {
    extern __shared__ __align__(16) unsigned short s_tile[];   // [6][132][64]
    const int b     = blockIdx.x >> 6;
    const int hbase = (blockIdx.x & 63) * 2;
    const int tid   = threadIdx.x;

    // ---- tile load: rows hbase-2 .. hbase+3, cols -2..129 (zero halo) ----
    for (int rt = 0; rt < 6; ++rt) {
        unsigned short* lr = s_tile + rt * (132 * 64);
        int gh = hbase - 2 + rt;
        if (gh >= 0 && gh < 128) {
            const unsigned short* gr = feat_t + (size_t)((b * 128 + gh) * 128) * 64;
            for (int u = tid; u < 512; u += 256)          // interior: 512 x 8 bf16 = first 64 cols * 64ch
                copy16_g2l(gr + u * 8, lr + 2 * 64 + u * 8);
            if (tid < 16) {                               // halo cols 0,1
                v4u z = {0, 0, 0, 0};
                *(v4u*)(lr + tid * 8) = z;
            } else if (tid < 32) {                        // halo cols 130,131
                v4u z = {0, 0, 0, 0};
                *(v4u*)(lr + 130 * 64 + (tid - 16) * 8) = z;
            }
        } else {
            for (int u = tid; u < 528; u += 256) {        // zero first 66 cols (528 x 8 bf16)
                v4u z = {0, 0, 0, 0};
                *(v4u*)(lr + u * 8) = z;
            }
        }
    }
    async_wait_all();
    __syncthreads();

    const int wave = tid >> 5, lane = tid & 31;
    const int row  = wave >> 2;                 // 0/1 -> output row hbase+row
    const int w0   = (wave & 3) * 32;
    const int nn   = lane & 15;
    const int h    = hbase + row;
    // lane-constant base: (tile_row=row, col=w0+nn, ch=khalf); taps add (r,c)
    const int laneBase = (row * 132 + w0 + nn) * 64 + (lane & 16);

    v8f acc[8] = {};                            // [mb 4][ntile 2]
    const v16bf* af = (const v16bf*)afrag;

    for (int r = 0; r < 5; ++r) {
#pragma unroll
        for (int c = 0; c < 5; ++c) {
#pragma unroll
            for (int q = 0; q < 2; ++q) {       // channel half of K=64
                const unsigned short* bp = s_tile + laneBase + (r * 132 + c) * 64 + q * 32;
                v16bf b0 = *(const v16bf*)bp;                 // N tile 0
                v16bf b1 = *(const v16bf*)(bp + 16 * 64);     // N tile 1 (+16 cols)
                const int ci = (r * 5 + c) * 2 + q;
                const v16bf* ab = af + (size_t)ci * 4 * 32 + lane;
#pragma unroll
                for (int mb = 0; mb < 4; ++mb) {
                    v16bf a = ab[mb * 32];
                    acc[mb * 2 + 0] = __builtin_amdgcn_wmma_f32_16x16x32_bf16(
                        false, a, false, b0, (short)0, acc[mb * 2 + 0], false, false);
                    acc[mb * 2 + 1] = __builtin_amdgcn_wmma_f32_16x16x32_bf16(
                        false, a, false, b1, (short)0, acc[mb * 2 + 1], false, false);
                }
            }
        }
    }

    // ---- store: D vgpr rr -> M = rr + (lane<16?0:8), N = lane%16 ----
    const int mofs = (lane & 16) ? 8 : 0;
#pragma unroll
    for (int mb = 0; mb < 4; ++mb)
#pragma unroll
        for (int ti = 0; ti < 2; ++ti) {
            v8f a = acc[mb * 2 + ti];
            int w = w0 + ti * 16 + nn;
#pragma unroll
            for (int rr = 0; rr < 8; ++rr) {
                int o = mb * 16 + rr + mofs;
                outp[((b * 64 + o) * 128 + h) * 128 + w] = a[rr] + fcb[o];
            }
        }
}

// ===========================================================================
// GDM kernels
// ===========================================================================
__global__ void k_pool3_mean(const float* __restrict__ x,
                             float* __restrict__ xp, float* __restrict__ ia) {
    int idx = blockIdx.x * blockDim.x + threadIdx.x;      // 8*256*256
    int b = idx >> 16, rem = idx & 65535;
    int h = rem >> 8, w = rem & 255;
    float acc = 0.f;
    for (int ic = 0; ic < 3; ++ic) {
        float s = 0.f;
        for (int dr = 0; dr < 3; ++dr) {
            int r = 2 * h - 1 + dr;
            if (r < 0 || r >= 512) continue;
            for (int dc = 0; dc < 3; ++dc) {
                int c = 2 * w - 1 + dc;
                if (c < 0 || c >= 512) continue;
                s += x[((b * 3 + ic) * 512 + r) * 512 + c];
            }
        }
        s *= (1.f / 9.f);
        xp[((b * 3 + ic) * 256 + h) * 256 + w] = s;
        acc += s;
    }
    ia[idx] = acc * (1.f / 3.f);
}

__global__ __launch_bounds__(256) void k_conv7_att(const float* __restrict__ xp,
                                                   const float* __restrict__ cw,
                                                   const float* __restrict__ gma,
                                                   const float* __restrict__ bta,
                                                   const float* __restrict__ mu,
                                                   const float* __restrict__ var,
                                                   float* __restrict__ att_sum) {
    int bo = blockIdx.x >> 6;               // 0..39 = b*5+oc
    int b = bo / 5, oc = bo % 5;
    int pix = (blockIdx.x & 63) * 256 + threadIdx.x;
    int i = pix >> 7, j = pix & 127;
    float s = 0.f;
    for (int ic = 0; ic < 3; ++ic)
        for (int r = 0; r < 7; ++r) {
            int ri = 2 * i + r - 3;
            if (ri < 0 || ri >= 256) continue;
            for (int c = 0; c < 7; ++c) {
                int cj = 2 * j + c - 3;
                if (cj < 0 || cj >= 256) continue;
                s += cw[((oc * 3 + ic) * 7 + r) * 7 + c] *
                     xp[((b * 3 + ic) * 256 + ri) * 256 + cj];
            }
        }
    float y = (s - mu[oc]) * rsqrtf(var[oc] + 1e-5f) * gma[oc] + bta[oc];
    __shared__ float red[256];
    red[threadIdx.x] = y;
    __syncthreads();
    for (int st = 128; st > 0; st >>= 1) {
        if (threadIdx.x < st) red[threadIdx.x] += red[threadIdx.x + st];
        __syncthreads();
    }
    if (threadIdx.x == 0) atomicAdd(&att_sum[bo], red[0]);
}

__global__ __launch_bounds__(256) void k_gl_hist(const float* __restrict__ ia,
                                                 const float* __restrict__ gw,
                                                 float* __restrict__ xgl,
                                                 int* __restrict__ xmax_bits,
                                                 unsigned* __restrict__ hist) {
    __shared__ unsigned lh[102];
    int tid = threadIdx.x;
    if (tid < 102) lh[tid] = 0u;
    __syncthreads();
    int b = blockIdx.y;
    int pix = blockIdx.x * 256 + tid;
    int h = pix >> 8, w = pix & 255;
    float s = 0.f;
    for (int r = 0; r < 5; ++r) {
        int ri = h + r - 2;
        if (ri < 0 || ri >= 256) continue;
        for (int c = 0; c < 5; ++c) {
            int ci = w + c - 2;
            if (ci < 0 || ci >= 256) continue;
            s += c_GL[r * 5 + c] * ia[(b << 16) + (ri << 8) + ci];
        }
    }
    s *= gw[0] * 4.f;                 // a1
    float v = fabsf(s);
    if (v < 1e-6f) v = 0.f;
    xgl[(b << 16) + pix] = v;
    if (v > 0.f) {
        atomicMax(&xmax_bits[b], __float_as_int(v));   // nonneg -> int cmp ok
        int t = (int)ceilf(100.f * cbrtf(v * 0.1f));
        if (t < 1) t = 1;
        if (t > 101) t = 101;
        while (t > 1 && v <= xval(t - 1)) --t;
        while (t <= 100 && v > xval(t)) ++t;
        atomicAdd(&lh[t], 1u);
    }
    __syncthreads();
    if (tid < 102 && lh[tid]) atomicAdd(&hist[b * 102 + tid], lh[tid]);
}

__global__ void k_small(const float* __restrict__ att_sum,
                        const unsigned* __restrict__ hist,
                        const int* __restrict__ xmax_bits,
                        float* __restrict__ thr,     // [8][4]
                        float* __restrict__ xmaxf) { // [8]
    int b = threadIdx.x;
    if (b >= 8) return;
    float att[5];
    for (int j = 0; j < 5; ++j) att[j] = tanhf(att_sum[b * 5 + j] * (1.f / 16384.f));
    float p[4] = {0.2f + 0.05f * att[1], 0.4f + 0.05f * att[2],
                  0.6f + 0.05f * att[3], 0.8f + 0.05f * att[4]};
    for (int i = 0; i < 3; ++i)
        for (int j = 0; j < 3 - i; ++j)
            if (p[j] > p[j + 1]) { float t = p[j]; p[j] = p[j + 1]; p[j + 1] = t; }
    unsigned all = 0;
    for (int t = 0; t < 102; ++t) all += hist[b * 102 + t];
    float inv = 1.f / (float)(all ? all : 1u);
    float best[4] = {3.4e38f, 3.4e38f, 3.4e38f, 3.4e38f};
    int bi[4] = {0, 0, 0, 0};
    unsigned run = 0;
    for (int t = 0; t <= 100; ++t) {
        run += hist[b * 102 + t];
        float pct = (float)run * inv;
        for (int k = 0; k < 4; ++k) {
            float z = fabsf(pct - p[k]);
            if (z < best[k]) { best[k] = z; bi[k] = t; }
        }
    }
    for (int k = 0; k < 4; ++k) thr[b * 4 + k] = xval(bi[k]);
    xmaxf[b] = __int_as_float(xmax_bits[b]);
}

__global__ void k_mask(const float* __restrict__ xgl,
                       const float* __restrict__ thr,
                       const float* __restrict__ xmaxf,
                       float* __restrict__ m) {
    int idx = blockIdx.x * blockDim.x + threadIdx.x;   // 8*65536
    int b = idx >> 16, pix = idx & 65535;
    float v = xgl[idx];
    float tm[6];
    tm[0] = 0.f;
    tm[1] = thr[b * 4 + 0]; tm[2] = thr[b * 4 + 1];
    tm[3] = thr[b * 4 + 2]; tm[4] = thr[b * 4 + 3];
    tm[5] = xmaxf[b];
    for (int q = 0; q < 5; ++q)
        m[((b * 5 + q) << 16) + pix] = (v > tm[q] && v <= tm[q + 1]) ? 1.f : 0.f;
}

__global__ void k_pool9(const float* __restrict__ in, float* __restrict__ out,
                        int stride, int outHW) {
    int idx = blockIdx.x * blockDim.x + threadIdx.x;
    int hw2 = outHW * outHW;
    int bq = idx / hw2, rem = idx % hw2;
    int i = rem / outHW, j = rem % outHW;
    float s = 0.f;
    int ib = i * stride - 4, jb = j * stride - 4;
    for (int r = 0; r < 9; ++r) {
        int ri = ib + r;
        if (ri < 0 || ri >= 256) continue;
        const float* row = in + bq * 65536 + ri * 256;
        for (int c = 0; c < 9; ++c) {
            int ci = jb + c;
            if (ci < 0 || ci >= 256) continue;
            s += row[ci];
        }
    }
    out[idx] = s * (1.f / 81.f);
}

__global__ __launch_bounds__(256) void k_norm(float* __restrict__ g) {
    int bq = blockIdx.x;
    float* p = g + bq * 16384;
    __shared__ float smn[256], smx[256];
    float mn = 3.4e38f, mx = -3.4e38f;
    for (int k = threadIdx.x; k < 16384; k += 256) {
        float v = p[k];
        mn = fminf(mn, v); mx = fmaxf(mx, v);
    }
    smn[threadIdx.x] = mn; smx[threadIdx.x] = mx;
    __syncthreads();
    for (int st = 128; st > 0; st >>= 1) {
        if (threadIdx.x < st) {
            smn[threadIdx.x] = fminf(smn[threadIdx.x], smn[threadIdx.x + st]);
            smx[threadIdx.x] = fmaxf(smx[threadIdx.x], smx[threadIdx.x + st]);
        }
        __syncthreads();
    }
    float MN = smn[0], MX = smx[0];
    float inv = 1.f / (MX - MN);
    for (int k = threadIdx.x; k < 16384; k += 256)
        p[k] = (p[k] - MN) * inv;
}

// ===========================================================================
extern "C" void kernel_launch(void* const* d_in, const int* in_sizes, int n_in,
                              void* d_out, int out_size, void* d_ws, size_t ws_size,
                              hipStream_t stream) {
    const float* x    = (const float*)d_in[0];   // [8,3,512,512]
    const float* gdmw = (const float*)d_in[1];   // [1,5]
    const float* cw   = (const float*)d_in[2];   // [5,3,7,7]
    const float* gma  = (const float*)d_in[3];
    const float* bta  = (const float*)d_in[4];
    const float* mu   = (const float*)d_in[5];
    const float* var  = (const float*)d_in[6];
    const float* feat = (const float*)d_in[7];   // [8,64,128,128]
    const float* hw   = (const float*)d_in[8];   // [64,64,3,3]
    const float* fcw  = (const float*)d_in[9];   // [64,320]
    const float* fcb  = (const float*)d_in[10];  // [64]

    float* gdm_out = (float*)d_out;                       // [8,5,128,128]
    float* hge_out = (float*)d_out + 8 * 5 * 128 * 128;   // [8,64,128,128]

    char* ws = (char*)d_ws;
    size_t off = 0;
    auto take = [&](size_t bytes) { char* p = ws + off; off = (off + bytes + 255) & ~(size_t)255; return p; };
    unsigned short* feat_t = (unsigned short*)take(8ull * 128 * 128 * 64 * 2);
    unsigned short* afrag  = (unsigned short*)take(50ull * 4 * 32 * 16 * 2);
    float*    Md       = (float*)take(5ull * 64 * 64 * 4);
    float*    xp       = (float*)take(8ull * 3 * 256 * 256 * 4);
    float*    ia       = (float*)take(8ull * 256 * 256 * 4);
    float*    xgl      = (float*)take(8ull * 256 * 256 * 4);
    float*    att_sum  = (float*)take(40 * 4);
    unsigned* hist     = (unsigned*)take(8 * 102 * 4);
    int*      xmax_b   = (int*)take(8 * 4);
    float*    thr      = (float*)take(8 * 4 * 4);
    float*    xmaxf    = (float*)take(8 * 4);
    float*    mbuf     = (float*)take(8ull * 5 * 256 * 256 * 4);
    float*    tbuf     = (float*)take(8ull * 5 * 256 * 256 * 4);
    (void)ws_size; (void)in_sizes; (void)n_in; (void)out_size;

    k_zero<<<1, 256, 0, stream>>>(att_sum, hist, xmax_b);

    // ---- HGE ----
    k_feat_t<<<8 * 128 * 2, 256, 0, stream>>>(feat, feat_t);
    k_prep<<<1, 256, 0, stream>>>(hw, fcw, Md, afrag);
    k_hge_gemm<<<8 * 64, 256, 6 * 132 * 64 * sizeof(unsigned short), stream>>>(
        feat_t, afrag, fcb, hge_out);

    // ---- GDM ----
    k_pool3_mean<<<8 * 256 * 256 / 256, 256, 0, stream>>>(x, xp, ia);
    k_conv7_att<<<40 * 64, 256, 0, stream>>>(xp, cw, gma, bta, mu, var, att_sum);
    k_gl_hist<<<dim3(256, 8), 256, 0, stream>>>(ia, gdmw, xgl, xmax_b, hist);
    k_small<<<1, 64, 0, stream>>>(att_sum, hist, xmax_b, thr, xmaxf);
    k_mask<<<8 * 65536 / 256, 256, 0, stream>>>(xgl, thr, xmaxf, mbuf);
    k_pool9<<<8 * 5 * 65536 / 256, 256, 0, stream>>>(mbuf, tbuf, 1, 256);
    k_pool9<<<8 * 5 * 65536 / 256, 256, 0, stream>>>(tbuf, mbuf, 1, 256);
    k_pool9<<<8 * 5 * 16384 / 256, 256, 0, stream>>>(mbuf, gdm_out, 2, 128);
    k_norm<<<40, 256, 0, stream>>>(gdm_out);
}
